// GraphSAGE_44839458570859
// MI455X (gfx1250) — matrix operand order repaired and checked by reading, not compiled, not run on hardware
//
#include <hip/hip_runtime.h>

#define NN   50000
#define EE   800000
#define K0   384
#define HH   256
#define GG   64

typedef __bf16 bf16x16 __attribute__((ext_vector_type(16)));
typedef float f32x8 __attribute__((ext_vector_type(8)));

struct Pack32 { uint4 lo, hi; };

__device__ __forceinline__ bf16x16 frag_from(uint4 lo, uint4 hi) {
    Pack32 p; p.lo = lo; p.hi = hi;
    return __builtin_bit_cast(bf16x16, p);
}

__device__ __forceinline__ unsigned short f2bf(float f) {
    unsigned int u = __float_as_uint(f);
    u += 0x7fffu + ((u >> 16) & 1u);          // round-to-nearest-even
    return (unsigned short)(u >> 16);
}
__device__ __forceinline__ float bf2f(unsigned int h) {
    return __uint_as_float(h << 16);
}

// ---------------------------------------------------------------------------
// Dual GEMM + bias:  C[m,n] = sum_k A1[m,k]*B1T[n,k] + sum_k A2[m,k]*B2T[n,k] + bias[n]
// A*: [nrows,K] bf16 row-major.  B*T: [HH,K] bf16 row-major (W transposed).
// Block = 8 waves = 128 rows x 64 cols. Both 64xK W tiles staged in LDS once
// (row stride padded to K+8 elements -> conflict-free ds_load_b128 frags).
// A fragments double-buffered in registers to overlap global latency w/ WMMA.
// ---------------------------------------------------------------------------
__global__ __launch_bounds__(256) void sage_gemm_wmma(
    const unsigned short* __restrict__ A1, const unsigned short* __restrict__ B1T,
    const unsigned short* __restrict__ A2, const unsigned short* __restrict__ B2T,
    const float* __restrict__ bias, float* __restrict__ C, int nrows, int K)
{
    __shared__ unsigned short sB[2 * 64 * (K0 + 8)];   // max-K sized; padded rows

    const int tid  = threadIdx.x;
    const int wave = tid >> 5;
    const int lane = tid & 31;
    const int lh   = lane >> 4;          // half-wave selects K sub-range
    const int ln   = lane & 15;
    const int rowBase = blockIdx.x * 128 + wave * 16;
    const int colBase = blockIdx.y * 64;
    const int stride  = K + 8;           // padded LDS row stride (elements)

    // ---- stage B1T / B2T 64-row tiles into LDS (once) ----
    {
        const int q    = K >> 3;         // uint4 per row
        const int totq = 2 * 64 * q;
        for (int i = tid; i < totq; i += 256) {
            int mat = i / (64 * q);
            int rem = i - mat * 64 * q;
            int col = rem / q;
            int kk  = (rem - col * q) << 3;
            const unsigned short* src =
                (mat ? B2T : B1T) + (size_t)(colBase + col) * K + kk;
            *(uint4*)&sB[(mat * 64 + col) * stride + kk] = *(const uint4*)src;
        }
    }
    __syncthreads();

    f32x8 acc[4];
#pragma unroll
    for (int t = 0; t < 4; ++t)
#pragma unroll
        for (int i = 0; i < 8; ++i) acc[t][i] = 0.0f;

    int r = rowBase + ln;                 // A row for this lane (M = lane&15)
    if (r >= nrows) r = nrows - 1;        // clamp: WMMA needs full EXEC
    const unsigned short* a1row = A1 + (size_t)r * K;
    const unsigned short* a2row = A2 + (size_t)r * K;

    // ---- phase 1: acc += A1 x B1 (B frags from LDS, A double-buffered) ----
    {
        uint4 alo = *(const uint4*)(a1row + lh * 8);
        uint4 ahi = *(const uint4*)(a1row + 16 + lh * 8);
        int k = 0;
        for (; k < K - 32; k += 32) {
            uint4 nlo = *(const uint4*)(a1row + k + 32 + lh * 8);
            uint4 nhi = *(const uint4*)(a1row + k + 48 + lh * 8);
            bf16x16 a = frag_from(alo, ahi);
#pragma unroll
            for (int t = 0; t < 4; ++t) {
                const unsigned short* bp = &sB[(t * 16 + ln) * stride + k + lh * 8];
                bf16x16 b = frag_from(*(const uint4*)bp, *(const uint4*)(bp + 16));
                acc[t] = __builtin_amdgcn_wmma_f32_16x16x32_bf16(
                    false, a, false, b, (short)0, acc[t], false, false);
            }
            alo = nlo; ahi = nhi;
        }
        bf16x16 a = frag_from(alo, ahi);
#pragma unroll
        for (int t = 0; t < 4; ++t) {
            const unsigned short* bp = &sB[(t * 16 + ln) * stride + k + lh * 8];
            bf16x16 b = frag_from(*(const uint4*)bp, *(const uint4*)(bp + 16));
            acc[t] = __builtin_amdgcn_wmma_f32_16x16x32_bf16(
                false, a, false, b, (short)0, acc[t], false, false);
        }
    }

    // ---- phase 2: acc += A2 x B2 ----
    {
        const unsigned short* sB2 = sB + 64 * stride;
        uint4 alo = *(const uint4*)(a2row + lh * 8);
        uint4 ahi = *(const uint4*)(a2row + 16 + lh * 8);
        int k = 0;
        for (; k < K - 32; k += 32) {
            uint4 nlo = *(const uint4*)(a2row + k + 32 + lh * 8);
            uint4 nhi = *(const uint4*)(a2row + k + 48 + lh * 8);
            bf16x16 a = frag_from(alo, ahi);
#pragma unroll
            for (int t = 0; t < 4; ++t) {
                const unsigned short* bp = &sB2[(t * 16 + ln) * stride + k + lh * 8];
                bf16x16 b = frag_from(*(const uint4*)bp, *(const uint4*)(bp + 16));
                acc[t] = __builtin_amdgcn_wmma_f32_16x16x32_bf16(
                    false, a, false, b, (short)0, acc[t], false, false);
            }
            alo = nlo; ahi = nhi;
        }
        bf16x16 a = frag_from(alo, ahi);
#pragma unroll
        for (int t = 0; t < 4; ++t) {
            const unsigned short* bp = &sB2[(t * 16 + ln) * stride + k + lh * 8];
            bf16x16 b = frag_from(*(const uint4*)bp, *(const uint4*)(bp + 16));
            acc[t] = __builtin_amdgcn_wmma_f32_16x16x32_bf16(
                false, a, false, b, (short)0, acc[t], false, false);
        }
    }

    // ---- epilogue: bias + store ----
#pragma unroll
    for (int t = 0; t < 4; ++t) {
        int col = colBase + t * 16 + ln;
        float bv = bias[col];
#pragma unroll
        for (int i = 0; i < 8; ++i) {
            int row = rowBase + lh * 8 + i;      // C layout: lanes16-31 hold M=8..15
            if (row < nrows) C[(size_t)row * HH + col] = acc[t][i] + bv;
        }
    }
}

// ---------------------------------------------------------------------------
// Utility / streaming kernels
// ---------------------------------------------------------------------------
__global__ void fill_f32_k(float* __restrict__ p, float v, int n) {
    int i = blockIdx.x * 256 + threadIdx.x;
    if (i < n) p[i] = v;
}

__global__ void wprep_k(const float* __restrict__ W, unsigned short* __restrict__ WT,
                        int K, int Hd) {
    int i = blockIdx.x * 256 + threadIdx.x;
    if (i >= K * Hd) return;
    int k = i / Hd, h = i - k * Hd;
    WT[(size_t)h * K + k] = f2bf(W[i]);          // WT[h][k] = W[k][h]
}

__global__ void concat_bf16_k(const float* __restrict__ g0, const float* __restrict__ g1,
                              const float* __restrict__ g2, unsigned short* __restrict__ x0,
                              int total) {
    int i = blockIdx.x * 256 + threadIdx.x;
    if (i >= total) return;
    int n = i / K0, c = i - n * K0;
    float v = (c < 128) ? g0[n * 128 + c]
            : (c < 256) ? g1[n * 128 + (c - 128)]
                        : g2[n * 128 + (c - 256)];
    x0[i] = f2bf(v);
}

__global__ void deg_k(const int* __restrict__ dst, float* __restrict__ deg, int e) {
    int i = blockIdx.x * 256 + threadIdx.x;
    if (i < e) atomicAdd(&deg[dst[i]], 1.0f);
}
__global__ void invdeg_k(const float* __restrict__ deg, float* __restrict__ inv, int n) {
    int i = blockIdx.x * 256 + threadIdx.x;
    if (i < n) inv[i] = 1.0f / fmaxf(deg[i], 1.0f);
}

// edge scatter: each thread handles one edge x 8 channels (bf16 read, f32 atomic accum)
__global__ void aggregate_k(const int* __restrict__ src, const int* __restrict__ dst,
                            const unsigned short* __restrict__ x, float* __restrict__ agg,
                            int total, int K) {
    int t = blockIdx.x * 256 + threadIdx.x;
    if (t >= total) return;
    int cpb = K >> 3;
    int e = t / cpb;
    int c = (t - e * cpb) << 3;
    int s = src[e], d = dst[e];
    uint4 p = *(const uint4*)(x + (size_t)s * K + c);
    float* o = agg + (size_t)d * K + c;
    atomicAdd(o + 0, bf2f(p.x & 0xffffu)); atomicAdd(o + 1, bf2f(p.x >> 16));
    atomicAdd(o + 2, bf2f(p.y & 0xffffu)); atomicAdd(o + 3, bf2f(p.y >> 16));
    atomicAdd(o + 4, bf2f(p.z & 0xffffu)); atomicAdd(o + 5, bf2f(p.z >> 16));
    atomicAdd(o + 6, bf2f(p.w & 0xffffu)); atomicAdd(o + 7, bf2f(p.w >> 16));
}

__global__ void scale_convert_k(const float* __restrict__ agg, const float* __restrict__ invd,
                                unsigned short* __restrict__ ab, int total, int K) {
    int i = blockIdx.x * 256 + threadIdx.x;
    if (i >= total) return;
    int n = i / K;
    ab[i] = f2bf(agg[i] * invd[n]);
}

// per-channel sum / sumsq (256 channels; 256 blocks stride over rows)
__global__ __launch_bounds__(256) void bn_stats_k(const float* __restrict__ hf,
                                                  float* __restrict__ stats, int nrows) {
    int ch = threadIdx.x;
    float s = 0.f, sq = 0.f;
    for (int r = blockIdx.x; r < nrows; r += gridDim.x) {
        float v = hf[(size_t)r * HH + ch];
        s += v; sq += v * v;
    }
    atomicAdd(&stats[ch], s);
    atomicAdd(&stats[HH + ch], sq);
}

__global__ void bn_apply_k(const float* __restrict__ hf, const float* __restrict__ stats,
                           const float* __restrict__ gamma, const float* __restrict__ beta,
                           unsigned short* __restrict__ xb, int nrows) {
    int i = blockIdx.x * 256 + threadIdx.x;
    if (i >= nrows * HH) return;
    int ch = i & (HH - 1);
    float invn = 1.0f / (float)nrows;
    float mu  = stats[ch] * invn;
    float var = stats[HH + ch] * invn - mu * mu;
    float v = (hf[i] - mu) * __frsqrt_rn(var + 1e-5f) * gamma[ch] + beta[ch];
    xb[i] = f2bf(fmaxf(v, 0.0f));
}

__global__ void pool_k(const unsigned short* __restrict__ x, const int* __restrict__ bid,
                       float* __restrict__ pooled, float* __restrict__ cnt, int total) {
    int t = blockIdx.x * 256 + threadIdx.x;
    if (t >= total) return;                       // total = NN * (HH/8)
    int n = t >> 5;
    int c = (t & 31) << 3;
    int g = bid[n];
    uint4 p = *(const uint4*)(x + (size_t)n * HH + c);
    float* o = pooled + (size_t)g * HH + c;
    atomicAdd(o + 0, bf2f(p.x & 0xffffu)); atomicAdd(o + 1, bf2f(p.x >> 16));
    atomicAdd(o + 2, bf2f(p.y & 0xffffu)); atomicAdd(o + 3, bf2f(p.y >> 16));
    atomicAdd(o + 4, bf2f(p.z & 0xffffu)); atomicAdd(o + 5, bf2f(p.z >> 16));
    atomicAdd(o + 6, bf2f(p.w & 0xffffu)); atomicAdd(o + 7, bf2f(p.w >> 16));
    if (c == 0) atomicAdd(&cnt[g], 1.0f);
}

// fc + relu + mlp head on [G, HH] -> [G, 2]; one block per graph
__global__ __launch_bounds__(256) void head_k(
    const float* __restrict__ pooled, const float* __restrict__ cnt,
    const float* __restrict__ fcW, const float* __restrict__ fcb,
    const float* __restrict__ W1, const float* __restrict__ b1,
    const float* __restrict__ W2, const float* __restrict__ b2,
    float* __restrict__ out)
{
    __shared__ float p[HH], y1[HH], y2[HH];
    int g = blockIdx.x, ch = threadIdx.x;
    float c = fmaxf(cnt[g], 1.0f);
    p[ch] = pooled[(size_t)g * HH + ch] / c;
    __syncthreads();
    float s = fcb[ch];
    for (int k = 0; k < HH; ++k) s += p[k] * fcW[k * HH + ch];
    y1[ch] = fmaxf(s, 0.0f);
    __syncthreads();
    s = b1[ch];
    for (int k = 0; k < HH; ++k) s += y1[k] * W1[k * HH + ch];
    y2[ch] = fmaxf(s, 0.0f);
    __syncthreads();
    if (ch < 2) {
        float s2 = b2[ch];
        for (int k = 0; k < HH; ++k) s2 += y2[k] * W2[k * 2 + ch];
        out[g * 2 + ch] = s2;
    }
}

#define CD(a, b) (((a) + (b) - 1) / (b))

extern "C" void kernel_launch(void* const* d_in, const int* in_sizes, int n_in,
                              void* d_out, int out_size, void* d_ws, size_t ws_size,
                              hipStream_t stream) {
    const float* g0   = (const float*)d_in[0];
    const float* g1   = (const float*)d_in[1];
    const float* g2   = (const float*)d_in[2];
    const int*   esrc = (const int*)d_in[3];
    const int*   edst = (const int*)d_in[4];
    const int*   bids = (const int*)d_in[5];
    const float* W0l  = (const float*)d_in[6];
    const float* W0r  = (const float*)d_in[7];
    const float* Wl   = (const float*)d_in[8];
    const float* Wr   = (const float*)d_in[9];
    const float* bconv= (const float*)d_in[10];
    const float* gam  = (const float*)d_in[11];
    const float* bet  = (const float*)d_in[12];
    const float* fcW  = (const float*)d_in[13];
    const float* fcb  = (const float*)d_in[14];
    const float* W1   = (const float*)d_in[15];
    const float* b1   = (const float*)d_in[16];
    const float* W2   = (const float*)d_in[17];
    const float* b2   = (const float*)d_in[18];
    float* out = (float*)d_out;

    char* base = (char*)d_ws;
    size_t cur = 0;
    auto carve = [&](size_t bytes) -> void* {
        cur = (cur + 255) & ~(size_t)255;
        void* p = base + cur;
        cur += bytes;
        return p;
    };
    unsigned short* x0b   = (unsigned short*)carve((size_t)NN * K0 * 2);
    unsigned short* xb    = (unsigned short*)carve((size_t)NN * HH * 2);
    float*          aggf  = (float*)carve((size_t)NN * K0 * 4);
    unsigned short* aggb  = (unsigned short*)carve((size_t)NN * K0 * 2);
    float*          hfb   = (float*)carve((size_t)NN * HH * 4);
    unsigned short* wlt0  = (unsigned short*)carve((size_t)HH * K0 * 2);
    unsigned short* wrt0  = (unsigned short*)carve((size_t)HH * K0 * 2);
    unsigned short* wlt   = (unsigned short*)carve((size_t)3 * HH * HH * 2);
    unsigned short* wrt   = (unsigned short*)carve((size_t)3 * HH * HH * 2);
    float*          degb  = (float*)carve((size_t)NN * 4);
    float*          invd  = (float*)carve((size_t)NN * 4);
    float*          stats = (float*)carve((size_t)2 * HH * 4);
    float*          pooled= (float*)carve((size_t)GG * HH * 4);
    float*          cntb  = (float*)carve((size_t)GG * 4);

    // weight prep: transpose + bf16 convert
    wprep_k<<<CD(K0 * HH, 256), 256, 0, stream>>>(W0l, wlt0, K0, HH);
    wprep_k<<<CD(K0 * HH, 256), 256, 0, stream>>>(W0r, wrt0, K0, HH);
    for (int i = 0; i < 3; ++i) {
        wprep_k<<<CD(HH * HH, 256), 256, 0, stream>>>(Wl + (size_t)i * HH * HH,
                                                      wlt + (size_t)i * HH * HH, HH, HH);
        wprep_k<<<CD(HH * HH, 256), 256, 0, stream>>>(Wr + (size_t)i * HH * HH,
                                                      wrt + (size_t)i * HH * HH, HH, HH);
    }

    // features: concat + bf16
    concat_bf16_k<<<CD(NN * K0, 256), 256, 0, stream>>>(g0, g1, g2, x0b, NN * K0);

    // degree
    fill_f32_k<<<CD(NN, 256), 256, 0, stream>>>(degb, 0.0f, NN);
    deg_k<<<CD(EE, 256), 256, 0, stream>>>(edst, degb, EE);
    invdeg_k<<<CD(NN, 256), 256, 0, stream>>>(degb, invd, NN);

    const unsigned short* xcur = x0b;
    int K = K0;
    for (int layer = 0; layer < 4; ++layer) {
        int elems = NN * K;
        fill_f32_k<<<CD(elems, 256), 256, 0, stream>>>(aggf, 0.0f, elems);
        int etot = EE * (K >> 3);
        aggregate_k<<<CD(etot, 256), 256, 0, stream>>>(esrc, edst, xcur, aggf, etot, K);
        scale_convert_k<<<CD(elems, 256), 256, 0, stream>>>(aggf, invd, aggb, elems, K);

        const unsigned short* wl_ = (layer == 0) ? wlt0 : wlt + (size_t)(layer - 1) * HH * HH;
        const unsigned short* wr_ = (layer == 0) ? wrt0 : wrt + (size_t)(layer - 1) * HH * HH;
        sage_gemm_wmma<<<dim3(CD(NN, 128), HH / 64), 256, 0, stream>>>(
            aggb, wl_, xcur, wr_, bconv + (size_t)layer * HH, hfb, NN, K);

        fill_f32_k<<<CD(2 * HH, 256), 256, 0, stream>>>(stats, 0.0f, 2 * HH);
        bn_stats_k<<<256, 256, 0, stream>>>(hfb, stats, NN);
        bn_apply_k<<<CD(NN * HH, 256), 256, 0, stream>>>(hfb, stats,
            gam + (size_t)layer * HH, bet + (size_t)layer * HH, xb, NN);

        xcur = xb;
        K = HH;
    }

    // pooling + head
    fill_f32_k<<<CD(GG * HH, 256), 256, 0, stream>>>(pooled, 0.0f, GG * HH);
    fill_f32_k<<<1, 256, 0, stream>>>(cntb, 0.0f, GG);
    pool_k<<<CD(NN * 32, 256), 256, 0, stream>>>(xb, bids, pooled, cntb, NN * 32);
    head_k<<<GG, 256, 0, stream>>>(pooled, cntb, fcW, fcb, W1, b1, W2, b2, out);
}
